// AttentionAggregator2d_35562329211815
// MI455X (gfx1250) — matrix operand order repaired and checked by reading, not compiled, not run on hardware
//
#include <hip/hip_runtime.h>

// ---------------------------------------------------------------------------
// AttentionAggregator2d for MI455X (gfx1250), wave32 + WMMA bf16 16x16x32.
// Fused two-pass softmax-attention: never materializes the 537MB attn matrix.
//   K1 pack_w : fuse Wq|Wk|Wv -> bf16 [320][256], biases -> f32 [320]
//   K2 proj   : WMMA GEMM [320,256]x[256,4096] per batch -> q,k (bf16 rows),
//               v (f32 [c][n])
//   K3 pass1  : WMMA S-tiles, online row max m_j and row sum Z_j
//   K4 vscale : vs[c][j] = bf16( v[c][j] / Z_j )   (folds 1/Z into V)
//   K5 pass2  : out[c,i] = sum_j vs[c,j]*exp(S[j,i]-m_j); out = zc + g*out
// ---------------------------------------------------------------------------

#define B_   8
#define CMC  256
#define CCC  256
#define PCH  32
#define NSP  4096          // 64*64
#define NT   (NSP / 16)    // 256 tiles

typedef __bf16 bf16_t;
typedef bf16_t bf16x16 __attribute__((ext_vector_type(16)));
typedef float  f32x8   __attribute__((ext_vector_type(8)));
typedef float  f32x4   __attribute__((ext_vector_type(4)));
typedef float  f32x2   __attribute__((ext_vector_type(2)));
typedef int    i32x4   __attribute__((ext_vector_type(4)));

union Frag16 { bf16x16 v; i32x4 q[2]; bf16_t h[16]; };
union Pack8  { bf16_t h[8]; i32x4 q; };

__device__ __forceinline__ bf16_t f2bf(float f) {
  unsigned u = __float_as_uint(f);
  u += 0x7FFFu + ((u >> 16) & 1u);          // round to nearest even
  unsigned short hv = (unsigned short)(u >> 16);
  bf16_t r; __builtin_memcpy(&r, &hv, sizeof(r));
  return r;
}

#define WMMA_BF16(a, b, c) \
  __builtin_amdgcn_wmma_f32_16x16x32_bf16(false, (a), false, (b), (short)0, (c), false, false)

// Workspace layout (bytes, 16B-aligned offsets)
static constexpr size_t WB_OFF   = 0;                                  // 320*256 bf16
static constexpr size_t BIAS_OFF = WB_OFF   + 320 * 256 * 2;           // 320 f32
static constexpr size_t QB_OFF   = BIAS_OFF + 320 * 4;                 // b*n*32 bf16
static constexpr size_t KB_OFF   = QB_OFF   + (size_t)B_ * NSP * PCH * 2;
static constexpr size_t VF_OFF   = KB_OFF   + (size_t)B_ * NSP * PCH * 2;   // b*c*n f32
static constexpr size_t VS_OFF   = VF_OFF   + (size_t)B_ * CCC * NSP * 4;   // b*c*n bf16
static constexpr size_t M_OFF    = VS_OFF   + (size_t)B_ * CCC * NSP * 2;   // b*n f32
static constexpr size_t RZ_OFF   = M_OFF    + (size_t)B_ * NSP * 4;
static constexpr size_t WS_TOTAL = RZ_OFF   + (size_t)B_ * NSP * 4;         // ~55 MB

// ---------------------------------------------------------------- K1: pack W
__global__ void pack_w_kernel(const float* __restrict__ Wq, const float* __restrict__ Wk,
                              const float* __restrict__ Wv, const float* __restrict__ bq,
                              const float* __restrict__ bk, const float* __restrict__ bv,
                              bf16_t* __restrict__ Wb, float* __restrict__ bias) {
  int idx = blockIdx.x * blockDim.x + threadIdx.x;
  if (idx >= 320 * CMC) return;
  int r = idx / CMC, c = idx % CMC;
  float w = (r < 32) ? Wq[r * CMC + c] : (r < 64) ? Wk[(r - 32) * CMC + c]
                                                  : Wv[(r - 64) * CMC + c];
  Wb[idx] = f2bf(w);
  if (c == 0)
    bias[r] = (r < 32) ? bq[r] : (r < 64) ? bk[r - 32] : bv[r - 64];
}

// -------------------------------------------------------- K2: WMMA projection
// grid (NT, 2, B), block 32 (one wave). Each wave: one 16-col n-tile,
// 10 M-tiles (half of 320 rows), K=256 in 8 steps of 32.
__global__ __launch_bounds__(32) void proj_kernel(
    const float* __restrict__ zm, const bf16_t* __restrict__ Wb,
    const float* __restrict__ bias, bf16_t* __restrict__ qb,
    bf16_t* __restrict__ kbuf, float* __restrict__ vf) {
  const int lane = threadIdx.x;
  const int col  = lane & 15;
  const int hi   = lane >> 4;
  const int n    = blockIdx.x * 16 + col;
  const int mh   = blockIdx.y;
  const int b    = blockIdx.z;

  const float* zmb = zm + (size_t)b * CMC * NSP;

  f32x8 acc[10];
#pragma unroll
  for (int t = 0; t < 10; ++t) {       // init C with bias (D row = v + 8*hi)
    const int mb_ = (mh * 10 + t) * 16 + 8 * hi;
    f32x4 b0 = *(const f32x4*)(bias + mb_);
    f32x4 b1 = *(const f32x4*)(bias + mb_ + 4);
    acc[t][0] = b0[0]; acc[t][1] = b0[1]; acc[t][2] = b0[2]; acc[t][3] = b0[3];
    acc[t][4] = b1[0]; acc[t][5] = b1[1]; acc[t][6] = b1[2]; acc[t][7] = b1[3];
  }

  for (int ks = 0; ks < 8; ++ks) {
    const int c0 = ks * 32;
    // B fragment: zm[c0..c0+31][n-tile]; lane half selects K 0-15 / 16-31
    Frag16 bfr;
    const int coff = c0 + hi * 16;
#pragma unroll
    for (int v = 0; v < 8; ++v) {
      float f0 = zmb[(size_t)(coff + 2 * v)     * NSP + n];
      float f1 = zmb[(size_t)(coff + 2 * v + 1) * NSP + n];
      bfr.h[2 * v]     = f2bf(f0);
      bfr.h[2 * v + 1] = f2bf(f1);
    }
#pragma unroll
    for (int t = 0; t < 10; ++t) {
      // A fragment: W row = M tile + col; K 16*hi .. 16*hi+15
      const bf16_t* wrow = Wb + (size_t)((mh * 10 + t) * 16 + col) * CMC;
      Frag16 af;
      af.q[0] = *(const i32x4*)(wrow + c0 + 16 * hi);
      af.q[1] = *(const i32x4*)(wrow + c0 + 16 * hi + 8);
      acc[t] = WMMA_BF16(af.v, bfr.v, acc[t]);
    }
  }

#pragma unroll
  for (int t = 0; t < 10; ++t) {
    const int mt = mh * 10 + t;
    const int rb = mt * 16 + 8 * hi;   // first of 8 consecutive output rows
    if (mt < 2) {                      // q rows 0..31 -> qb[b][n][32] bf16
      Pack8 p;
#pragma unroll
      for (int v = 0; v < 8; ++v) p.h[v] = f2bf(acc[t][v]);
      *(i32x4*)(qb + ((size_t)b * NSP + n) * PCH + rb) = p.q;
    } else if (mt < 4) {               // k rows 32..63 -> kbuf[b][n][32] bf16
      Pack8 p;
#pragma unroll
      for (int v = 0; v < 8; ++v) p.h[v] = f2bf(acc[t][v]);
      *(i32x4*)(kbuf + ((size_t)b * NSP + n) * PCH + (rb - 32)) = p.q;
    } else {                           // v rows 64..319 -> vf[b][c][n] f32
#pragma unroll
      for (int v = 0; v < 8; ++v)
        vf[((size_t)b * CCC + (rb - 64 + v)) * NSP + n] = acc[t][v];
    }
  }
}

// ----------------------------------------- K3: row stats m_j, 1/Z_j (pass 1)
// grid (NT, B), block 32. Online softmax stats along row j over all i.
__global__ __launch_bounds__(32) void pass1_kernel(
    const bf16_t* __restrict__ qb, const bf16_t* __restrict__ kbuf,
    float* __restrict__ mrow, float* __restrict__ rzrow) {
  const int lane = threadIdx.x;
  const int col  = lane & 15;
  const int hi   = lane >> 4;
  const int j0   = blockIdx.x * 16;
  const int b    = blockIdx.y;

  const bf16_t* qrow = qb + ((size_t)b * NSP + j0 + col) * PCH;
  Frag16 af;
  af.q[0] = *(const i32x4*)(qrow + 8 * hi);
  af.q[1] = *(const i32x4*)(qrow + 16 + 8 * hi);

  const bf16_t* kbb = kbuf + (size_t)b * NSP * PCH;
  float m[8], Z[8];
#pragma unroll
  for (int v = 0; v < 8; ++v) { m[v] = -1e30f; Z[v] = 0.0f; }
  const f32x8 zero8 = {};

  for (int i0 = 0; i0 < NSP; i0 += 16) {
    const bf16_t* krow = kbb + (size_t)(i0 + col) * PCH + hi * 8;
    Frag16 bfr;
    bfr.q[0] = *(const i32x4*)(krow);
    bfr.q[1] = *(const i32x4*)(krow + 16);
    f32x8 s = WMMA_BF16(af.v, bfr.v, zero8);
#pragma unroll
    for (int v = 0; v < 8; ++v) {
      float sv = s[v];
      float nm = fmaxf(m[v], sv);
      Z[v] = Z[v] * __expf(m[v] - nm) + __expf(sv - nm);
      m[v] = nm;
    }
  }
  // merge partial (m,Z) across the 16 lanes holding the same rows
#pragma unroll
  for (int mask = 1; mask <= 8; mask <<= 1) {
#pragma unroll
    for (int v = 0; v < 8; ++v) {
      float mo = __shfl_xor(m[v], mask, 32);
      float Zo = __shfl_xor(Z[v], mask, 32);
      float nm = fmaxf(m[v], mo);
      Z[v] = Z[v] * __expf(m[v] - nm) + Zo * __expf(mo - nm);
      m[v] = nm;
    }
  }
  if (col == 0) {        // lanes 0 (rows 0-7) and 16 (rows 8-15) write
#pragma unroll
    for (int v = 0; v < 8; ++v) {
      int j = j0 + 8 * hi + v;
      mrow[(size_t)b * NSP + j]  = m[v];
      rzrow[(size_t)b * NSP + j] = 1.0f / Z[v];
    }
  }
}

// --------------------------------------------------- K4: vs = bf16(v * 1/Z_j)
__global__ void vscale_kernel(const float* __restrict__ vf,
                              const float* __restrict__ rz,
                              bf16_t* __restrict__ vs) {
  int idx = blockIdx.x * blockDim.x + threadIdx.x;   // B*CC*N = 8<<20
  int j = idx & (NSP - 1);
  int b = idx >> 20;                                  // CC*N = 1<<20
  vs[idx] = f2bf(vf[idx] * rz[(size_t)b * NSP + j]);
}

// ------------------------------------------------ K5: out = zc + g * (V @ P)
// grid (NT, B), block 32. One wave: 16-col i-tile, all 16 c-tiles (full C=256),
// loop j in steps of 32: 2 S-WMMAs -> exp -> LDS transpose -> B-frag -> 16
// out-WMMAs.
__global__ __launch_bounds__(32) void pass2_kernel(
    const bf16_t* __restrict__ qb, const bf16_t* __restrict__ kbuf,
    const bf16_t* __restrict__ vs, const float* __restrict__ mrow,
    const float* __restrict__ zc, const float* __restrict__ gam,
    float* __restrict__ out) {
  __shared__ float pl[16 * 36];      // [i][j] f32, stride 36 kills bank conflicts
  const int lane = threadIdx.x;
  const int col  = lane & 15;
  const int hi   = lane >> 4;
  const int i0   = blockIdx.x * 16;
  const int b    = blockIdx.y;

  // k B-fragment for this i-tile is loop-invariant
  const bf16_t* krow = kbuf + ((size_t)b * NSP + i0 + col) * PCH + hi * 8;
  Frag16 kf;
  kf.q[0] = *(const i32x4*)(krow);
  kf.q[1] = *(const i32x4*)(krow + 16);

  f32x8 acc[16];
#pragma unroll
  for (int t = 0; t < 16; ++t) acc[t] = (f32x8){};
  const f32x8 zero8 = {};

  const bf16_t* qbb = qb + (size_t)b * NSP * PCH;
  const bf16_t* vsb = vs + (size_t)b * CCC * NSP;
  const float*  mb  = mrow + (size_t)b * NSP;

  for (int jb = 0; jb < NSP; jb += 32) {
    // S tiles for j rows jb..jb+15 and jb+16..jb+31
    const bf16_t* q0 = qbb + (size_t)(jb + col) * PCH;
    const bf16_t* q1 = qbb + (size_t)(jb + 16 + col) * PCH;
    Frag16 a0, a1;
    a0.q[0] = *(const i32x4*)(q0 + 8 * hi);
    a0.q[1] = *(const i32x4*)(q0 + 16 + 8 * hi);
    a1.q[0] = *(const i32x4*)(q1 + 8 * hi);
    a1.q[1] = *(const i32x4*)(q1 + 16 + 8 * hi);
    f32x8 s0 = WMMA_BF16(a0.v, kf.v, zero8);
    f32x8 s1 = WMMA_BF16(a1.v, kf.v, zero8);

    // per-row m_j for the D-layout rows of each tile
    f32x4 m0a = *(const f32x4*)(mb + jb + 8 * hi);
    f32x4 m0b = *(const f32x4*)(mb + jb + 8 * hi + 4);
    f32x4 m1a = *(const f32x4*)(mb + jb + 16 + 8 * hi);
    f32x4 m1b = *(const f32x4*)(mb + jb + 16 + 8 * hi + 4);

    f32x4 p0a, p0b, p1a, p1b;
#pragma unroll
    for (int v = 0; v < 4; ++v) {
      p0a[v] = __expf(s0[v]     - m0a[v]);
      p0b[v] = __expf(s0[v + 4] - m0b[v]);
      p1a[v] = __expf(s1[v]     - m1a[v]);
      p1b[v] = __expf(s1[v + 4] - m1b[v]);
    }

    // D layout -> LDS [i][j] (i = col, j = tile_off + 8*hi + v)
    float* lp = pl + col * 36;
    *(f32x4*)(lp + 8 * hi)          = p0a;
    *(f32x4*)(lp + 8 * hi + 4)      = p0b;
    *(f32x4*)(lp + 16 + 8 * hi)     = p1a;
    *(f32x4*)(lp + 16 + 8 * hi + 4) = p1b;
    __syncthreads();

    // read back as 32x16 bf16 B-fragment: lane=N(i), VGPR v = K pair (2v,2v+1)
    Frag16 pf;
    const float* rp = pl + col * 36 + hi * 16;
#pragma unroll
    for (int v = 0; v < 8; ++v) {
      f32x2 t = *(const f32x2*)(rp + 2 * v);
      pf.h[2 * v]     = f2bf(t[0]);
      pf.h[2 * v + 1] = f2bf(t[1]);
    }

    // out += V(A) x P(B) for all 16 c-tiles
#pragma unroll
    for (int ct = 0; ct < 16; ++ct) {
      const bf16_t* vrow = vsb + (size_t)(ct * 16 + col) * NSP + jb;
      Frag16 vfg;
      vfg.q[0] = *(const i32x4*)(vrow + 16 * hi);
      vfg.q[1] = *(const i32x4*)(vrow + 16 * hi + 8);
      acc[ct] = WMMA_BF16(vfg.v, pf.v, acc[ct]);
    }
    __syncthreads();
  }

  const float g = gam[0];
  const int i = i0 + col;
#pragma unroll
  for (int ct = 0; ct < 16; ++ct) {
#pragma unroll
    for (int v = 0; v < 8; ++v) {
      int c = ct * 16 + 8 * hi + v;
      size_t idx = ((size_t)b * CCC + c) * NSP + i;
      out[idx] = zc[idx] + g * acc[ct][v];
    }
  }
}

// ---------------------------------------------------------------------------
extern "C" void kernel_launch(void* const* d_in, const int* in_sizes, int n_in,
                              void* d_out, int out_size, void* d_ws, size_t ws_size,
                              hipStream_t stream) {
  (void)in_sizes; (void)n_in; (void)out_size;
  const float* zc    = (const float*)d_in[0];
  const float* zm    = (const float*)d_in[1];
  const float* Wq    = (const float*)d_in[2];
  const float* bq    = (const float*)d_in[3];
  const float* Wk    = (const float*)d_in[4];
  const float* bk    = (const float*)d_in[5];
  const float* Wv    = (const float*)d_in[6];
  const float* bv    = (const float*)d_in[7];
  const float* gamma = (const float*)d_in[8];
  float* out = (float*)d_out;

  if (ws_size < WS_TOTAL) return;
  char* ws = (char*)d_ws;
  bf16_t* Wb   = (bf16_t*)(ws + WB_OFF);
  float*  bias = (float*) (ws + BIAS_OFF);
  bf16_t* qb   = (bf16_t*)(ws + QB_OFF);
  bf16_t* kb   = (bf16_t*)(ws + KB_OFF);
  float*  vf   = (float*) (ws + VF_OFF);
  bf16_t* vs   = (bf16_t*)(ws + VS_OFF);
  float*  mrow = (float*) (ws + M_OFF);
  float*  rz   = (float*) (ws + RZ_OFF);

  pack_w_kernel<<<(320 * CMC + 255) / 256, 256, 0, stream>>>(Wq, Wk, Wv, bq, bk, bv, Wb, bias);
  proj_kernel<<<dim3(NT, 2, B_), 32, 0, stream>>>(zm, Wb, bias, qb, kb, vf);
  pass1_kernel<<<dim3(NT, B_), 32, 0, stream>>>(qb, kb, mrow, rz);
  vscale_kernel<<<(B_ * CCC * NSP) / 256, 256, 0, stream>>>(vf, rz, vs);
  pass2_kernel<<<dim3(NT, B_), 32, 0, stream>>>(qb, kb, vs, mrow, zc, gamma, out);
}
